// PatchMAE_9414568313202
// MI455X (gfx1250) — matrix operand order repaired and checked by reading, not compiled, not run on hardware
//
#include <hip/hip_runtime.h>
#include <hip/hip_bf16.h>
#include <math.h>

// ---------------------------------------------------------------------------
// PatchMAE forward for MI455X (gfx1250, wave32, WMMA, async-LDS staging).
// f16 operand mirrors + transposed f16 weights -> all WMMA fragments are
// contiguous b128 loads. Attention stages V band chunks into LDS with
// global_load_async_to_lds_b128 (ASYNCcnt).
// ---------------------------------------------------------------------------

typedef __attribute__((ext_vector_type(16))) _Float16 v16h;
typedef __attribute__((ext_vector_type(8)))  _Float16 v8h;
typedef __attribute__((ext_vector_type(8)))  float    v8f;

#define BB    4
#define SEQ   1024
#define NVv   32
#define PLh   16
#define Dm    512
#define Hh    8
#define ELAY  4
#define DFF   2048
#define Pp    64
#define Ll    2048
#define DH    64
#define MROWS (BB*Ll)       // 8192
#define SCL   0.125f

#define WMMA_F16(A_, B_, C_) \
  __builtin_amdgcn_wmma_f32_16x16x32_f16(false, (A_), false, (B_), (short)0, (C_), false, false)

// A-fragment k index (ISA 7.12.2): runs of 8 contiguous k's.
__device__ __forceinline__ int fkB(int lane, int e) {   // B frag: k = g*16 + e
  return (((lane >> 4) & 1) << 4) + e;
}

// Build a v16h fragment from two contiguous 16-byte runs.
__device__ __forceinline__ v16h ld2(const _Float16* p0, const _Float16* p1) {
  v8h lo = *(const v8h*)p0;
  v8h hi = *(const v8h*)p1;
  v16h r;
  #pragma unroll
  for (int e = 0; e < 8; ++e) { r[e] = lo[e]; r[e + 8] = hi[e]; }
  return r;
}

// ---------------------------------------------------------------- stats ----
__global__ void stats_kernel(const float* __restrict__ x, float* __restrict__ means,
                             float* __restrict__ stds) {
  int pair = blockIdx.x;           // b*32+nv
  int b = pair >> 5, nv = pair & 31;
  int tid = threadIdx.x;           // 256
  float s = 0.f, sq = 0.f;
  for (int t = tid; t < SEQ; t += 256) {
    float v = x[((size_t)b * SEQ + t) * NVv + nv];
    s += v; sq += v * v;
  }
  __shared__ float shs[256], shq[256];
  shs[tid] = s; shq[tid] = sq;
  __syncthreads();
  for (int off = 128; off > 0; off >>= 1) {
    if (tid < off) { shs[tid] += shs[tid + off]; shq[tid] += shq[tid + off]; }
    __syncthreads();
  }
  if (tid == 0) {
    float mean = shs[0] * (1.f / SEQ);
    float var  = shq[0] * (1.f / SEQ) - mean * mean;
    means[pair] = mean;
    stds[pair]  = sqrtf(var + 1e-5f);
  }
}

// ---------------------------------------------------- weight pack (f16^T) --
// Wt[n*K + k] = (f16) W[k*N + n]
__global__ void pack_kernel(const float* __restrict__ W, _Float16* __restrict__ Wt,
                            int K, int N) {
  int idx = blockIdx.x * 256 + threadIdx.x;
  if (idx >= K * N) return;
  int n = idx / K, k = idx - n * K;
  Wt[idx] = (_Float16)W[(size_t)k * N + n];
}

// ---------------------------------------------------------------- embed ----
__global__ void embed_kernel(const float* __restrict__ x, const unsigned char* __restrict__ mask,
                             const float* __restrict__ We, const float* __restrict__ be,
                             const float* __restrict__ means, const float* __restrict__ stds,
                             float* __restrict__ tok, _Float16* __restrict__ tokH) {
  int row = blockIdx.x;                 // ((b*P)+p)*NV + nv
  int b = row >> 11, rem = row & 2047, p = rem >> 5, nv = rem & 31;
  int tid = threadIdx.x;                // 128
  __shared__ float sp[PLh];
  if (tid < PLh) {
    float xv = x[((size_t)b * SEQ + p * PLh + tid) * NVv + nv];
    sp[tid] = (xv - means[b * NVv + nv]) / stds[b * NVv + nv];
  }
  __syncthreads();
  bool msk = mask[((b * Pp) + p) * NVv + nv] != 0;
  for (int d = tid; d < Dm; d += 128) {
    float s = be[d];
    #pragma unroll
    for (int t = 0; t < PLh; ++t) s += sp[t] * We[t * Dm + d];
    float v = msk ? 0.f : s;
    tok[(size_t)row * Dm + d]  = v;
    tokH[(size_t)row * Dm + d] = (_Float16)v;
  }
}

// ----------------------------------------------------------- WMMA GEMM ----
// C = act(Ah[M,K] @ Wt[N,K]^T + bias); one wave per 16x(16*JT) tile.
// All fragment loads are contiguous b128s.
template <int JT>
__global__ void gemm_kernel(const _Float16* __restrict__ Ah, const _Float16* __restrict__ Wt,
                            const float* __restrict__ bias, float* __restrict__ Cf,
                            _Float16* __restrict__ Ch, int M, int N, int K, int gelu) {
  int lane = threadIdx.x;                    // 32
  int g = (lane >> 4) & 1;
  int tileM = blockIdx.y << 4;
  int tileN = blockIdx.x * (JT * 16);
  int mrow = tileM + (lane & 15);
  const _Float16* arow = Ah + (size_t)mrow * K;

  v8f acc[JT];
  #pragma unroll
  for (int j = 0; j < JT; ++j) { v8f z = {}; acc[j] = z; }

  for (int k0 = 0; k0 < K; k0 += 32) {
    v16h a = ld2(arow + k0 + g * 8, arow + k0 + 16 + g * 8);
    #pragma unroll
    for (int j = 0; j < JT; ++j) {
      const _Float16* wrow = Wt + (size_t)(tileN + j * 16 + (lane & 15)) * K + k0 + g * 8;
      v16h bf = ld2(wrow, wrow + 16);
      acc[j] = WMMA_F16(a, bf, acc[j]);
    }
  }
  int rb = g << 3;
  #pragma unroll
  for (int j = 0; j < JT; ++j) {
    int ncol = tileN + j * 16 + (lane & 15);
    float bc = bias[ncol];
    #pragma unroll
    for (int r = 0; r < 8; ++r) {
      float v = acc[j][r] + bc;
      if (gelu) v = 0.5f * v * (1.f + erff(v * 0.70710678118f));
      size_t off = (size_t)(tileM + rb + r) * N + ncol;
      if (Cf) Cf[off] = v;
      if (Ch) Ch[off] = (_Float16)v;
    }
  }
}

// ------------------------------------------------------ banded attention ---
// Block = (p, h, b), 2 waves x 16 query rows. Band = patches [p-4, p+4].
// Q/K frags: b128 from f16 mirrors. V: async-staged to LDS per 32-row chunk.
__global__ void attn_kernel(const _Float16* __restrict__ q, const _Float16* __restrict__ k,
                            const _Float16* __restrict__ v, _Float16* __restrict__ o) {
  int p = blockIdx.x, h = blockIdx.y, b = blockIdx.z;
  int wave = threadIdx.x >> 5;
  int lane = threadIdx.x & 31;
  int g = (lane >> 4) & 1;
  __shared__ __align__(16) float    sS[2][16][288];
  __shared__ __align__(16) _Float16 sP[2][16][288];
  __shared__ __align__(16) _Float16 sV[32][64];
  int p0 = p - 4; if (p0 < 0) p0 = 0;
  int p1 = p + 4; if (p1 > Pp - 1) p1 = Pp - 1;
  int nk = (p1 - p0 + 1) * NVv;             // multiple of 32
  int hb = h * DH;
  size_t rowBase = (size_t)b * Ll;

  // Q fragments
  const _Float16* qrow = q + (rowBase + p * NVv + wave * 16 + (lane & 15)) * Dm + hb;
  v16h aq0 = ld2(qrow + g * 8,      qrow + 16 + g * 8);
  v16h aq1 = ld2(qrow + 32 + g * 8, qrow + 48 + g * 8);

  // scores = (Q @ K^T) * scale over the band
  int tTiles = nk >> 4;
  for (int t = 0; t < tTiles; ++t) {
    const _Float16* krow = k + (rowBase + p0 * NVv + t * 16 + (lane & 15)) * Dm + hb;
    v16h b0 = ld2(krow + g * 8,      krow + 16 + g * 8);
    v16h b1 = ld2(krow + 32 + g * 8, krow + 48 + g * 8);
    v8f c = {};
    c = WMMA_F16(aq0, b0, c);
    c = WMMA_F16(aq1, b1, c);
    int rb = g << 3;
    #pragma unroll
    for (int r = 0; r < 8; ++r)
      sS[wave][rb + r][t * 16 + (lane & 15)] = c[r] * SCL;
  }
  __syncthreads();

  // row softmax (lanes 0..15, one row each)
  if (lane < 16) {
    float mx = -1e30f;
    for (int c = 0; c < nk; ++c) mx = fmaxf(mx, sS[wave][lane][c]);
    float sum = 0.f;
    for (int c = 0; c < nk; ++c) {
      float e = __expf(sS[wave][lane][c] - mx);
      sS[wave][lane][c] = e;
      sum += e;
    }
    float inv = 1.f / sum;
    for (int c = 0; c < nk; ++c)
      sP[wave][lane][c] = (_Float16)(sS[wave][lane][c] * inv);
  }
  __syncthreads();

  unsigned sVoff = (unsigned)(size_t)&sV[0][0];   // LDS byte offset (addr[31:0])

  // O = P @ V, V chunks async-staged into LDS
  v8f o0 = {}, o1 = {}, o2 = {}, o3 = {};
  int chunks = nk >> 5;
  for (int c = 0; c < chunks; ++c) {
    int vr0 = p0 * NVv + c * 32;
    __syncthreads();
    {
      const _Float16* vbase = v + (rowBase + vr0) * Dm + hb;
      int t64 = threadIdx.x;                       // 0..63
      #pragma unroll
      for (int i = 0; i < 4; ++i) {
        int chk = i * 64 + t64;                    // 0..255 16B segments
        int r = chk >> 3, seg = chk & 7;
        unsigned lds = sVoff + (unsigned)(r * 128 + seg * 16);
        unsigned long long ga =
            (unsigned long long)(size_t)(vbase + (size_t)r * Dm) + (unsigned)(seg * 16);
        asm volatile("global_load_async_to_lds_b128 %0, %1, off"
                     :: "v"(lds), "v"(ga) : "memory");
      }
      asm volatile("s_wait_asynccnt 0x0" ::: "memory");
    }
    __syncthreads();

    v16h a = ld2(&sP[wave][lane & 15][c * 32 + g * 16],
                 &sP[wave][lane & 15][c * 32 + g * 16 + 8]);
    v16h bb;
    #pragma unroll
    for (int e = 0; e < 16; ++e) bb[e] = sV[fkB(lane, e)][0  + (lane & 15)];
    o0 = WMMA_F16(a, bb, o0);
    #pragma unroll
    for (int e = 0; e < 16; ++e) bb[e] = sV[fkB(lane, e)][16 + (lane & 15)];
    o1 = WMMA_F16(a, bb, o1);
    #pragma unroll
    for (int e = 0; e < 16; ++e) bb[e] = sV[fkB(lane, e)][32 + (lane & 15)];
    o2 = WMMA_F16(a, bb, o2);
    #pragma unroll
    for (int e = 0; e < 16; ++e) bb[e] = sV[fkB(lane, e)][48 + (lane & 15)];
    o3 = WMMA_F16(a, bb, o3);
  }

  int rb = g << 3;
  int orow = p * NVv + wave * 16 + rb;
  #pragma unroll
  for (int r = 0; r < 8; ++r) {
    size_t base = (rowBase + orow + r) * Dm + hb + (lane & 15);
    o[base + 0]  = (_Float16)o0[r];
    o[base + 16] = (_Float16)o1[r];
    o[base + 32] = (_Float16)o2[r];
    o[base + 48] = (_Float16)o3[r];
  }
}

// ------------------------------------------------ residual + layernorm -----
__global__ void addln_kernel(const float* __restrict__ in, const float* __restrict__ res,
                             float* __restrict__ out, _Float16* __restrict__ outH,
                             const float* __restrict__ g, const float* __restrict__ bt) {
  int row = blockIdx.x;
  int lane = threadIdx.x;                 // 32
  float xv[16];
  float s = 0.f, sq = 0.f;
  #pragma unroll
  for (int i = 0; i < 16; ++i) {
    int c = lane + i * 32;
    float v = in[(size_t)row * Dm + c];
    if (res) v += res[(size_t)row * Dm + c];
    xv[i] = v; s += v; sq += v * v;
  }
  #pragma unroll
  for (int off = 16; off > 0; off >>= 1) {
    s  += __shfl_xor(s, off, 32);
    sq += __shfl_xor(sq, off, 32);
  }
  float mean = s * (1.f / Dm);
  float var  = sq * (1.f / Dm) - mean * mean;
  float rstd = rsqrtf(var + 1e-5f);
  #pragma unroll
  for (int i = 0; i < 16; ++i) {
    int c = lane + i * 32;
    float v = (xv[i] - mean) * rstd * g[c] + bt[c];
    out[(size_t)row * Dm + c]  = v;
    outH[(size_t)row * Dm + c] = (_Float16)v;
  }
}

// ------------------------------------------------------------- loss --------
__global__ void loss_kernel(const float* __restrict__ recon, const float* __restrict__ x,
                            const unsigned char* __restrict__ mask,
                            const float* __restrict__ means, const float* __restrict__ stds,
                            float* accum) {
  int idx = blockIdx.x * 256 + threadIdx.x;
  int row = idx >> 4, t = idx & 15;
  int b = row >> 11, rem = row & 2047, p = rem >> 5, nv = rem & 31;
  float num = 0.f, cnt = 0.f;
  if (mask[((b * Pp) + p) * NVv + nv]) {
    float sc = stds[b * NVv + nv], bi = means[b * NVv + nv];
    float rec  = recon[(size_t)row * PLh + t] * sc + bi;
    float targ = x[((size_t)b * SEQ + p * PLh + t) * NVv + nv];
    float d = rec - targ;
    num = d * d; cnt = 1.f;
  }
  __shared__ float shn[256], shc[256];
  int tid = threadIdx.x;
  shn[tid] = num; shc[tid] = cnt;
  __syncthreads();
  for (int off = 128; off > 0; off >>= 1) {
    if (tid < off) { shn[tid] += shn[tid + off]; shc[tid] += shc[tid + off]; }
    __syncthreads();
  }
  if (tid == 0) {
    atomicAdd(&accum[0], shn[0]);
    atomicAdd(&accum[1], shc[0]);
  }
}

__global__ void finalize_kernel(const float* accum, float* out) {
  float den = accum[1];
  out[0] = accum[0] / (den > 0.f ? den : 1.f);
}

// ------------------------------------------------------------ launch -------
extern "C" void kernel_launch(void* const* d_in, const int* in_sizes, int n_in,
                              void* d_out, int out_size, void* d_ws, size_t ws_size,
                              hipStream_t stream) {
  const float*         x_enc  = (const float*)d_in[0];
  const unsigned char* mask   = (const unsigned char*)d_in[1];
  const float* W_embed = (const float*)d_in[2];
  const float* b_embed = (const float*)d_in[3];
  const float* Wq = (const float*)d_in[4];  const float* bq = (const float*)d_in[5];
  const float* Wk = (const float*)d_in[6];  const float* bk = (const float*)d_in[7];
  const float* Wv = (const float*)d_in[8];  const float* bv = (const float*)d_in[9];
  const float* Wo = (const float*)d_in[10]; const float* bo = (const float*)d_in[11];
  const float* W1 = (const float*)d_in[12]; const float* b1 = (const float*)d_in[13];
  const float* W2 = (const float*)d_in[14]; const float* b2 = (const float*)d_in[15];
  const float* g1 = (const float*)d_in[16]; const float* be1 = (const float*)d_in[17];
  const float* g2 = (const float*)d_in[18]; const float* be2 = (const float*)d_in[19];
  const float* gf = (const float*)d_in[20]; const float* bf = (const float*)d_in[21];
  const float* W_proj = (const float*)d_in[22];
  const float* b_proj = (const float*)d_in[23];

  // ---- workspace carve-up (all 256B aligned) ----
  char* pw = (char*)d_ws;
  auto alloc = [&](size_t bytes) -> char* {
    char* r = pw; pw += (bytes + 255) & ~(size_t)255; return r;
  };
  const size_t AF = (size_t)MROWS * Dm;            // activation elements
  float*    tok   = (float*)alloc(AF * 4);
  float*    bufE  = (float*)alloc(AF * 4);         // attn_out / FFN2 out / final-LN f32
  float*    recon = (float*)alloc((size_t)MROWS * PLh * 4);
  float*    means = (float*)alloc(BB * NVv * 4);
  float*    stds  = (float*)alloc(BB * NVv * 4);
  float*    accum = (float*)alloc(2 * 4);
  _Float16* tokH  = (_Float16*)alloc(AF * 2);
  _Float16* qH    = (_Float16*)alloc(AF * 2);      // also reused as encH at the end
  _Float16* kH    = (_Float16*)alloc(AF * 2);
  _Float16* vH    = (_Float16*)alloc(AF * 2);
  _Float16* oH    = (_Float16*)alloc(AF * 2);
  _Float16* y1H   = (_Float16*)alloc((size_t)MROWS * DFF * 2);
  _Float16* WtQ   = (_Float16*)alloc((size_t)ELAY * Dm * Dm * 2);
  _Float16* WtK   = (_Float16*)alloc((size_t)ELAY * Dm * Dm * 2);
  _Float16* WtV   = (_Float16*)alloc((size_t)ELAY * Dm * Dm * 2);
  _Float16* WtO   = (_Float16*)alloc((size_t)ELAY * Dm * Dm * 2);
  _Float16* Wt1   = (_Float16*)alloc((size_t)ELAY * Dm * DFF * 2);
  _Float16* Wt2   = (_Float16*)alloc((size_t)ELAY * DFF * Dm * 2);
  _Float16* WtP   = (_Float16*)alloc((size_t)Dm * PLh * 2);

  hipMemsetAsync(accum, 0, 2 * sizeof(float), stream);

  // ---- one-time weight packing (f16, transposed [N][K]) ----
  int nDD = (Dm * Dm + 255) / 256, nDF = (Dm * DFF + 255) / 256;
  for (int l = 0; l < ELAY; ++l) {
    size_t sDD = (size_t)l * Dm * Dm, sDF = (size_t)l * Dm * DFF;
    pack_kernel<<<nDD, 256, 0, stream>>>(Wq + sDD, WtQ + sDD, Dm, Dm);
    pack_kernel<<<nDD, 256, 0, stream>>>(Wk + sDD, WtK + sDD, Dm, Dm);
    pack_kernel<<<nDD, 256, 0, stream>>>(Wv + sDD, WtV + sDD, Dm, Dm);
    pack_kernel<<<nDD, 256, 0, stream>>>(Wo + sDD, WtO + sDD, Dm, Dm);
    pack_kernel<<<nDF, 256, 0, stream>>>(W1 + sDF, Wt1 + sDF, Dm, DFF);   // K=512,N=2048
    pack_kernel<<<nDF, 256, 0, stream>>>(W2 + sDF, Wt2 + sDF, DFF, Dm);   // K=2048,N=512
  }
  pack_kernel<<<(Dm * PLh + 255) / 256, 256, 0, stream>>>(W_proj, WtP, Dm, PLh);

  stats_kernel<<<BB * NVv, 256, 0, stream>>>(x_enc, means, stds);
  embed_kernel<<<MROWS, 128, 0, stream>>>(x_enc, mask, W_embed, b_embed, means, stds, tok, tokH);

  dim3 gD(Dm / 64, MROWS / 16);     // JT=4, N=512
  dim3 gF(DFF / 64, MROWS / 16);    // JT=4, N=2048
  dim3 gP(1, MROWS / 16);           // JT=1, N=16
  dim3 gAttn(Pp, Hh, BB);

  for (int l = 0; l < ELAY; ++l) {
    size_t sDD = (size_t)l * Dm * Dm, sDF = (size_t)l * Dm * DFF;
    const float* bql = bq + (size_t)l * Dm;   const float* bkl = bk + (size_t)l * Dm;
    const float* bvl = bv + (size_t)l * Dm;   const float* bol = bo + (size_t)l * Dm;
    const float* b1l = b1 + (size_t)l * DFF;  const float* b2l = b2 + (size_t)l * Dm;

    gemm_kernel<4><<<gD, 32, 0, stream>>>(tokH, WtQ + sDD, bql, nullptr, qH, MROWS, Dm, Dm, 0);
    gemm_kernel<4><<<gD, 32, 0, stream>>>(tokH, WtK + sDD, bkl, nullptr, kH, MROWS, Dm, Dm, 0);
    gemm_kernel<4><<<gD, 32, 0, stream>>>(tokH, WtV + sDD, bvl, nullptr, vH, MROWS, Dm, Dm, 0);
    attn_kernel<<<gAttn, 64, 0, stream>>>(qH, kH, vH, oH);
    gemm_kernel<4><<<gD, 32, 0, stream>>>(oH, WtO + sDD, bol, bufE, nullptr, MROWS, Dm, Dm, 0);
    addln_kernel<<<MROWS, 32, 0, stream>>>(tok, bufE, tok, tokH,
                                           g1 + (size_t)l * Dm, be1 + (size_t)l * Dm);
    gemm_kernel<4><<<gF, 32, 0, stream>>>(tokH, Wt1 + sDF, b1l, nullptr, y1H, MROWS, DFF, Dm, 1);
    gemm_kernel<4><<<gD, 32, 0, stream>>>(y1H, Wt2 + sDF, b2l, bufE, nullptr, MROWS, Dm, DFF, 0);
    addln_kernel<<<MROWS, 32, 0, stream>>>(tok, bufE, tok, tokH,
                                           g2 + (size_t)l * Dm, be2 + (size_t)l * Dm);
  }

  addln_kernel<<<MROWS, 32, 0, stream>>>(tok, nullptr, bufE, qH /*encH*/, gf, bf);
  gemm_kernel<1><<<gP, 32, 0, stream>>>(qH, WtP, b_proj, recon, nullptr, MROWS, PLh, Dm, 0);
  loss_kernel<<<(MROWS * PLh) / 256, 256, 0, stream>>>(recon, x_enc, mask, means, stds, accum);
  finalize_kernel<<<1, 1, 0, stream>>>(accum, (float*)d_out);
}
